// VanillaVQ_1657857376702
// MI455X (gfx1250) — matrix operand order repaired and checked by reading, not compiled, not run on hardware
//
#include <hip/hip_runtime.h>
#include <hip/hip_bf16.h>
#include <math.h>

// VQ-VAE vector quantization for MI455X (gfx1250, wave32, WMMA).
// Fused 32768x8192x128 GEMM + row argmin on the bf16 WMMA pipe with bf16
// hi/lo error compensation; codebook chunks double-buffered through LDS via
// CDNA5 async global->LDS DMA (ASYNCcnt) so staging overlaps the WMMAs.

#define CODEBOOK 8192
#define EMBED    128
#define NVEC     32768
#define BETA     0.25f

typedef __attribute__((ext_vector_type(16))) __bf16 v16bf;
typedef __attribute__((ext_vector_type(8)))  float  v8f;

union ABreg { v16bf v; unsigned u[8]; uint4 q[2]; };

__device__ __forceinline__ unsigned bf16_rne(float x) {
  unsigned u = __float_as_uint(x);
  return (u + 0x7FFFu + ((u >> 16) & 1u)) >> 16;   // bf16 bits, round-nearest-even
}
__device__ __forceinline__ float bf16f(unsigned h) { return __uint_as_float(h << 16); }

// ---- CDNA5 async global->LDS copy (ASYNCcnt) with synchronous fallback -----
#if defined(__has_builtin)
#if __has_builtin(__builtin_amdgcn_global_load_async_to_lds_b128) && \
    __has_builtin(__builtin_amdgcn_s_wait_asynccnt)
#define VQ_ASYNC 1
#endif
#endif

#ifdef VQ_ASYNC
// Builtin signature (from hipcc diagnostic): (v4i AS1*, v4i AS3*, imm, imm)
typedef int v4i __attribute__((vector_size(16)));
typedef __attribute__((address_space(1))) v4i* g_v4i_p;   // global pointee
typedef __attribute__((address_space(3))) v4i* l_v4i_p;   // LDS pointee
#endif

__device__ __forceinline__ void async_b128(const uint4* g, uint4* l) {
#ifdef VQ_ASYNC
  __builtin_amdgcn_global_load_async_to_lds_b128((g_v4i_p)g, (l_v4i_p)l, 0, 0);
#else
  *l = *g;
#endif
}

#ifdef VQ_ASYNC
#define WAIT_ASYNC(n) __builtin_amdgcn_s_wait_asynccnt(n)
#else
#define WAIT_ASYNC(n) ((void)0)
#endif

// ---------------------------------------------------------------------------
// Zero the histogram (harness does not re-zero ws between replays).
__global__ void vq_zero(unsigned* __restrict__ counts) {
  int i = blockIdx.x * blockDim.x + threadIdx.x;
  if (i < CODEBOOK) counts[i] = 0u;
}

// ---------------------------------------------------------------------------
// Split codebook into bf16 hi/lo and compute |e_k|^2 in fp32.
__global__ __launch_bounds__(256) void vq_prep(const float* __restrict__ emb,
                                               unsigned* __restrict__ eh,
                                               unsigned* __restrict__ el,
                                               float* __restrict__ e2) {
  const int wave = threadIdx.x >> 5, lane = threadIdx.x & 31;
  const int row  = blockIdx.x * 8 + wave;
  float4 v = *(const float4*)(emb + (size_t)row * EMBED + lane * 4);
  float xs[4] = {v.x, v.y, v.z, v.w};
  unsigned h[4], l[4];
  float ss = 0.f;
#pragma unroll
  for (int i = 0; i < 4; ++i) {
    h[i] = bf16_rne(xs[i]);
    l[i] = bf16_rne(xs[i] - bf16f(h[i]));
    ss  += xs[i] * xs[i];
  }
  uint2 ph = make_uint2(h[0] | (h[1] << 16), h[2] | (h[3] << 16));
  uint2 pl = make_uint2(l[0] | (l[1] << 16), l[2] | (l[3] << 16));
  ((uint2*)eh)[row * 32 + lane] = ph;   // row-major: 64 uints (128 bf16) per code
  ((uint2*)el)[row * 32 + lane] = pl;
#pragma unroll
  for (int m = 16; m >= 1; m >>= 1) ss += __shfl_xor(ss, m, 32);
  if (lane == 0) e2[row] = ss;
}

// ---------------------------------------------------------------------------
// Stage one 64-column codebook chunk (hi+lo) into an LDS buffer.
// 2048 uint4 total -> 8 async b128 per thread (= 8 ASYNCcnt per wave).
__device__ __forceinline__ void stage_chunk(const uint4* __restrict__ ehq,
                                            const uint4* __restrict__ elq,
                                            unsigned* bufH, unsigned* bufL,
                                            int tid, int cb) {
  const uint4* srcH = ehq + cb * 16;
  const uint4* srcL = elq + cb * 16;
#pragma unroll
  for (int i = 0; i < 4; ++i) {
    const int j   = tid + i * 256;             // 1024 uint4 per matrix
    const int dst = (j >> 4) * 17 + (j & 15);  // padded col stride = 17 uint4
    async_b128(srcH + j, (uint4*)bufH + dst);
    async_b128(srcL + j, (uint4*)bufL + dst);
  }
}

// ---------------------------------------------------------------------------
// Fused WMMA GEMM + argmin. One WG = 128 rows (8 waves x 16-row A tiles held
// in VGPRs). Codebook streamed in 64-column chunks, double-buffered in LDS.
// Score = |e|^2 - 2*(z.e); |z|^2 dropped (row-constant).
__global__ __launch_bounds__(256) void vq_argmin(const float*    __restrict__ z,
                                                 const unsigned* __restrict__ eh,
                                                 const unsigned* __restrict__ el,
                                                 const float*    __restrict__ e2,
                                                 unsigned*       __restrict__ idx_out) {
  // 64 columns x 128 bf16 (=64 uints), padded to 68 uints/col vs. bank
  // conflicts; two buffers for copy/compute overlap. 68 KB of 320 KB WGP LDS.
  __shared__ unsigned sEH[2][64 * 68];
  __shared__ unsigned sEL[2][64 * 68];

  const int tid  = threadIdx.x;
  const int wave = tid >> 5, lane = tid & 31;
  const int kh   = lane >> 4, l16 = lane & 15;   // lane half / lane-in-half
  const int m0   = blockIdx.x * 128 + wave * 16;

  // ---- Load this wave's 16x128 A tile, split to bf16 hi/lo, WMMA A layout
  // (16-bit A 16x32, ISA 7.12.2: lane halves hold complementary K groups).
  ABreg ah[4], al[4];
  {
    const float* zr = z + (size_t)(m0 + l16) * EMBED;
#pragma unroll
    for (int ks = 0; ks < 4; ++ks) {
      const int g0 = ks * 32 + kh * 16;
      float4 a0 = *(const float4*)(zr + g0);
      float4 a1 = *(const float4*)(zr + g0 + 4);
      float4 b0 = *(const float4*)(zr + g0 + 8);
      float4 b1 = *(const float4*)(zr + g0 + 12);
      float xs[16] = {a0.x,a0.y,a0.z,a0.w, a1.x,a1.y,a1.z,a1.w,
                      b0.x,b0.y,b0.z,b0.w, b1.x,b1.y,b1.z,b1.w};
#pragma unroll
      for (int i = 0; i < 8; ++i) {
        float x0 = xs[2*i], x1 = xs[2*i+1];
        unsigned h0 = bf16_rne(x0), h1 = bf16_rne(x1);
        ah[ks].u[i] = h0 | (h1 << 16);
        unsigned q0 = bf16_rne(x0 - bf16f(h0));
        unsigned q1 = bf16_rne(x1 - bf16f(h1));
        al[ks].u[i] = q0 | (q1 << 16);
      }
    }
  }

  float    minv[8];
  unsigned mini[8];
#pragma unroll
  for (int r = 0; r < 8; ++r) { minv[r] = 3.0e38f; mini[r] = 0u; }

  const uint4* ehq = (const uint4*)eh;
  const uint4* elq = (const uint4*)el;

  // Prefetch chunk 0 into buffer 0.
  stage_chunk(ehq, elq, sEH[0], sEL[0], tid, 0);

  for (int cb = 0; cb < CODEBOOK; cb += 64) {
    const int buf = (cb >> 6) & 1;
    if (cb + 64 < CODEBOOK) {
      // Overlap: kick off next chunk into the other buffer, then wait only
      // for the current buffer's 8 per-wave async ops (in-order completion).
      stage_chunk(ehq, elq, sEH[buf ^ 1], sEL[buf ^ 1], tid, cb + 64);
      WAIT_ASYNC(8);
    } else {
      WAIT_ASYNC(0);
    }
    __syncthreads();   // all waves' staging of `buf` now visible

    const unsigned* bH = sEH[buf];
    const unsigned* bL = sEL[buf];
#pragma unroll
    for (int sub = 0; sub < 4; ++sub) {
      const int cs = sub * 16 + l16;             // this lane's column in chunk
      v8f c = {};
#pragma unroll
      for (int ks = 0; ks < 4; ++ks) {
        // B layout (32x16 bf16): lane = column, lanes 0-15 K 32ks..+15,
        // lanes 16-31 K 32ks+16..+31 (contiguous k order, same as A).
        const unsigned* bpH = bH + cs * 68 + ks * 16 + kh * 8;
        const unsigned* bpL = bL + cs * 68 + ks * 16 + kh * 8;
        ABreg bh, bl;
        bh.q[0] = *(const uint4*)bpH; bh.q[1] = *(const uint4*)(bpH + 4);
        bl.q[0] = *(const uint4*)bpL; bl.q[1] = *(const uint4*)(bpL + 4);
        // Error-compensated product: zh*eh + zl*eh + zh*el (zl*el ~ 2^-18)
        c = __builtin_amdgcn_wmma_f32_16x16x32_bf16(false, ah[ks].v, false, bh.v,
                                                    (short)0, c, false, false);
        c = __builtin_amdgcn_wmma_f32_16x16x32_bf16(false, al[ks].v, false, bh.v,
                                                    (short)0, c, false, false);
        c = __builtin_amdgcn_wmma_f32_16x16x32_bf16(false, ah[ks].v, false, bl.v,
                                                    (short)0, c, false, false);
      }
      const int   colg = cb + sub * 16 + l16;
      const float ek2  = e2[colg];
      // C layout: VGPR r holds row r (lanes 0-15) / row r+8 (lanes 16-31).
#pragma unroll
      for (int r = 0; r < 8; ++r) {
        float s = ek2 - 2.0f * c[r];
        if (s < minv[r]) { minv[r] = s; mini[r] = (unsigned)colg; }
      }
    }
    __syncthreads();   // everyone done reading `buf` before it is re-staged
  }

  // Reduce across the 16 column-lanes; masks 1..8 keep lane halves separate.
#pragma unroll
  for (int m = 1; m < 16; m <<= 1) {
#pragma unroll
    for (int r = 0; r < 8; ++r) {
      float    ov = __shfl_xor(minv[r], m, 32);
      unsigned oi = (unsigned)__shfl_xor((int)mini[r], m, 32);
      if (ov < minv[r] || (ov == minv[r] && oi < mini[r])) { minv[r] = ov; mini[r] = oi; }
    }
  }
  if (l16 == 0) {                      // lane 0 -> rows m0..+7, lane 16 -> +8..+15
    const int base = m0 + kh * 8;
#pragma unroll
    for (int r = 0; r < 8; ++r) idx_out[base + r] = mini[r];
  }
}

// ---------------------------------------------------------------------------
// Gather z_q = emb[idx], write output, per-block SSE partial, histogram.
__global__ __launch_bounds__(256) void vq_gather(const float*    __restrict__ z,
                                                 const float*    __restrict__ emb,
                                                 const unsigned* __restrict__ idx,
                                                 float*          __restrict__ zq,
                                                 unsigned*       __restrict__ counts,
                                                 float*          __restrict__ partial) {
  __shared__ float ps[8];
  const int wave = threadIdx.x >> 5, lane = threadIdx.x & 31;
  const int row  = blockIdx.x * 8 + wave;
  const unsigned id = idx[row];
  float4 ev = ((const float4*)(emb + (size_t)id  * EMBED))[lane];
  float4 zv = ((const float4*)(z   + (size_t)row * EMBED))[lane];
  ((float4*)(zq + (size_t)row * EMBED))[lane] = ev;   // z_q_st == z_q in value
  float dx = ev.x - zv.x, dy = ev.y - zv.y, dz = ev.z - zv.z, dw = ev.w - zv.w;
  float ss = dx*dx + dy*dy + dz*dz + dw*dw;
#pragma unroll
  for (int m = 16; m >= 1; m >>= 1) ss += __shfl_xor(ss, m, 32);
  if (lane == 0) { ps[wave] = ss; atomicAdd(&counts[id], 1u); }
  __syncthreads();
  if (threadIdx.x == 0) {
    float s = 0.f;
#pragma unroll
    for (int i = 0; i < 8; ++i) s += ps[i];
    partial[blockIdx.x] = s;   // fixed-order -> deterministic
  }
}

// ---------------------------------------------------------------------------
// Single block: deterministic final reductions -> commit_loss, perplexity.
__global__ __launch_bounds__(256) void vq_final(const float*    __restrict__ partial,
                                                const unsigned* __restrict__ counts,
                                                float*          __restrict__ out2) {
  __shared__ float red[256];
  __shared__ float sLoss;
  const int t = threadIdx.x;
  float s = 0.f;
  for (int i = t; i < 4096; i += 256) s += partial[i];
  red[t] = s;
  __syncthreads();
  for (int w = 128; w > 0; w >>= 1) { if (t < w) red[t] += red[t + w]; __syncthreads(); }
  if (t == 0) sLoss = red[0] * (1.0f + BETA) / 4194304.0f;  // (beta+1)*mean((zq-z)^2)
  __syncthreads();
  float hsum = 0.f;
  for (int i = t; i < CODEBOOK; i += 256) {
    float em = (float)counts[i] * (1.0f / 32768.0f);
    hsum += em * logf(em + 1e-8f);
  }
  red[t] = hsum;
  __syncthreads();
  for (int w = 128; w > 0; w >>= 1) { if (t < w) red[t] += red[t + w]; __syncthreads(); }
  if (t == 0) { out2[0] = sLoss; out2[1] = expf(-red[0]); }
}

// ---------------------------------------------------------------------------
extern "C" void kernel_launch(void* const* d_in, const int* in_sizes, int n_in,
                              void* d_out, int out_size, void* d_ws, size_t ws_size,
                              hipStream_t stream) {
  (void)in_sizes; (void)n_in; (void)out_size; (void)ws_size;
  const float* z   = (const float*)d_in[0];   // [8,4096,128] fp32
  const float* emb = (const float*)d_in[1];   // [8192,128]   fp32
  float* out = (float*)d_out;                 // 4194304 zq + loss + perplexity

  // Workspace layout (~4.4 MB total)
  unsigned* eh      = (unsigned*)d_ws;                 // 8192*64 u32 = 2 MB (bf16 hi)
  unsigned* el      = eh + CODEBOOK * 64;              // 2 MB (bf16 lo)
  float*    e2      = (float*)(el + CODEBOOK * 64);    // 32 KB
  unsigned* idx     = (unsigned*)(e2 + CODEBOOK);      // 128 KB
  unsigned* counts  = idx + NVEC;                      // 32 KB
  float*    partial = (float*)(counts + CODEBOOK);     // 16 KB (4096 blocks)

  vq_zero  <<<32,   256, 0, stream>>>(counts);
  vq_prep  <<<1024, 256, 0, stream>>>(emb, eh, el, e2);
  vq_argmin<<<256,  256, 0, stream>>>(z, eh, el, e2, idx);
  vq_gather<<<4096, 256, 0, stream>>>(z, emb, idx, out, counts, partial);
  vq_final <<<1,    256, 0, stream>>>(partial, counts, out + (size_t)NVEC * EMBED);
}
